// SS2DBidirectionalScan_13142599926109
// MI455X (gfx1250) — matrix operand order repaired and checked by reading, not compiled, not run on hardware
//
#include <hip/hip_runtime.h>
#include <hip/hip_bf16.h>
#include <cstddef>

// ---------------- problem constants ----------------
#define Bsz   8
#define Cch   128
#define Lseq  4096          // 64*64
#define DI    160           // D_INNER
#define DS    16            // D_STATE
#define RK    8             // DT_RANK
#define XDBL_LD 48          // 40 padded to 48 (multiple of 16)
#define Mrows (Bsz*Lseq)    // 32768

typedef float  v8f_t  __attribute__((ext_vector_type(8)));
typedef __bf16 v16bf_t __attribute__((ext_vector_type(16)));

union Frag16 { v16bf_t v; uint4 q[2]; };

__device__ __forceinline__ unsigned short f2bf(float f) {
    unsigned u = __float_as_uint(f);
    u += 0x7FFFu + ((u >> 16) & 1u);           // round-to-nearest-even
    return (unsigned short)(u >> 16);
}
__device__ __forceinline__ float bf2f(unsigned short h) {
    return __uint_as_float((unsigned)h << 16);
}
__device__ __forceinline__ void store1(float* p, float v)          { *p = v; }
__device__ __forceinline__ void store1(unsigned short* p, float v) { *p = f2bf(v); }

// ---------------- fp32 -> bf16 weight convert with row padding ----------------
__global__ void convert_pad_bf16(const float* __restrict__ src, unsigned short* __restrict__ dst,
                                 int rows, int cols, int dstRows) {
    int idx = blockIdx.x * 256 + threadIdx.x;
    int total = dstRows * cols;
    if (idx >= total) return;
    int r = idx / cols;
    dst[idx] = (r < rows) ? f2bf(src[idx]) : (unsigned short)0;
}

// ---------------- layernorm over C=128 + transpose (B,C,L)->(B*L,128) bf16 ----------------
__global__ __launch_bounds__(256) void layernorm_kernel(const float* __restrict__ x,
                                                        const float* __restrict__ g,
                                                        const float* __restrict__ be,
                                                        unsigned short* __restrict__ seqn) {
    const int lane = threadIdx.x & 31, wave = threadIdx.x >> 5;
    const int row  = blockIdx.x * 8 + wave;          // b*Lseq + l
    const int b = row >> 12, l = row & (Lseq - 1);
    const float* xb = x + (size_t)b * Cch * Lseq + l;
    float v[4], s = 0.f, sq = 0.f;
#pragma unroll
    for (int i = 0; i < 4; ++i) {
        int c = lane + i * 32;
        v[i] = xb[(size_t)c * Lseq];
        s += v[i]; sq += v[i] * v[i];
    }
#pragma unroll
    for (int o = 16; o; o >>= 1) { s += __shfl_xor(s, o, 32); sq += __shfl_xor(sq, o, 32); }
    const float mean = s * (1.f / 128.f);
    const float var  = sq * (1.f / 128.f) - mean * mean;
    const float inv  = rsqrtf(var + 1e-5f);
#pragma unroll
    for (int i = 0; i < 4; ++i) {
        int c = lane + i * 32;
        seqn[(size_t)row * Cch + c] = f2bf((v[i] - mean) * inv * g[c] + be[c]);
    }
}

// ---------------- register-blocked bf16 WMMA GEMM ----------------
// C[M, N(+off)] = A[M,K] * W[N,K]^T.  Each wave owns an MT x NT grid of 16x16 tiles:
// A fragments are reused across NT column tiles and B fragments across MT row tiles,
// so each K-step issues MT*NT v_wmma for only 2*(MT+NT) b128 loads (vs 4 loads / 1 wmma
// unblocked). Working set is L2-resident (192 MB) so fragments stream from L2, no LDS.
template <typename OutT, int MT, int NT>
__global__ __launch_bounds__(256) void gemm_wmma(const unsigned short* __restrict__ A,
                                                 const unsigned short* __restrict__ W,
                                                 OutT* __restrict__ Co,
                                                 int K, int lda, int ldc, int ncol_off, int rev) {
    const int lane  = threadIdx.x & 31;
    const int wave  = threadIdx.x >> 5;
    const int tileM = blockIdx.y * (8 * MT * 16) + wave * (MT * 16);
    const int tileN = blockIdx.x * (NT * 16);
    const int khalf = (lane >> 4) * 8;    // A: two 8-elem contiguous K chunks per lane
    const int kq    = (lane >> 4) * 8;    // B: two 8-elem contiguous K chunks per lane

    const unsigned short* Arow[MT];
#pragma unroll
    for (int mt = 0; mt < MT; ++mt) {
        int m = tileM + mt * 16 + (lane & 15);
        if (rev) m = (m & ~(Lseq - 1)) | ((Lseq - 1) - (m & (Lseq - 1)));   // reversed rows
        Arow[mt] = A + (size_t)m * lda;
    }
    const unsigned short* Wrow[NT];
#pragma unroll
    for (int nt = 0; nt < NT; ++nt)
        Wrow[nt] = W + (size_t)(tileN + nt * 16 + (lane & 15)) * K;

    v8f_t acc[MT][NT];
#pragma unroll
    for (int mt = 0; mt < MT; ++mt)
#pragma unroll
        for (int nt = 0; nt < NT; ++nt) acc[mt][nt] = (v8f_t){};

    for (int k0 = 0; k0 < K; k0 += 32) {
        Frag16 a[MT], b[NT];
#pragma unroll
        for (int mt = 0; mt < MT; ++mt) {
            if (k0 + 32 < K) __builtin_prefetch(Arow[mt] + k0 + 32 + khalf, 0, 1); // global_prefetch_b8
            a[mt].q[0] = *reinterpret_cast<const uint4*>(Arow[mt] + k0 + khalf);
            a[mt].q[1] = *reinterpret_cast<const uint4*>(Arow[mt] + k0 + khalf + 16);
        }
#pragma unroll
        for (int nt = 0; nt < NT; ++nt) {
            b[nt].q[0] = *reinterpret_cast<const uint4*>(Wrow[nt] + k0 + kq);
            b[nt].q[1] = *reinterpret_cast<const uint4*>(Wrow[nt] + k0 + kq + 16);
        }
#pragma unroll
        for (int mt = 0; mt < MT; ++mt)
#pragma unroll
            for (int nt = 0; nt < NT; ++nt)
                acc[mt][nt] = __builtin_amdgcn_wmma_f32_16x16x32_bf16(
                    false, a[mt].v, false, b[nt].v, (short)0, acc[mt][nt], false, false);
    }

    // D layout: VGPR r -> M = r + 8*(lane>=16), N = lane%16
#pragma unroll
    for (int mt = 0; mt < MT; ++mt) {
        const int mst = tileM + mt * 16 + (lane >> 4) * 8;
#pragma unroll
        for (int nt = 0; nt < NT; ++nt) {
            const int nst = tileN + nt * 16 + (lane & 15) + ncol_off;
#pragma unroll
            for (int r = 0; r < 8; ++r)
                store1(Co + (size_t)(mst + r) * ldc + nst, acc[mt][nt][r]);
        }
    }
}

// ---------------- depthwise causal conv(4) + SiLU ----------------
__global__ __launch_bounds__(256) void conv_silu_kernel(const unsigned short* __restrict__ xz,
                                                        const unsigned short* __restrict__ cw,
                                                        const float* __restrict__ cb,
                                                        unsigned short* __restrict__ xmc) {
    int idx = blockIdx.x * 256 + threadIdx.x;           // over B*L*DI
    if (idx >= Bsz * Lseq * DI) return;
    const int d  = idx % DI;
    const int bl = idx / DI;
    const int l  = bl & (Lseq - 1);
    const int b  = bl >> 12;
    float acc = cb[d];
#pragma unroll
    for (int k = 0; k < 4; ++k) {
        int ll = l + k - 3;
        if (ll >= 0)
            acc += bf2f(cw[d * 4 + k]) * bf2f(xz[((size_t)b * Lseq + ll) * (2 * DI) + d]);
    }
    float s = acc / (1.f + __expf(-acc));               // SiLU
    xmc[idx] = f2bf(s);
}

// ---------------- selective scan: lane = (d-half, n); h = 1 reg/lane ----------------
__global__ __launch_bounds__(256) void scan_kernel(const float* __restrict__ xdbl,
                                                   const unsigned short* __restrict__ xmc,
                                                   const unsigned short* __restrict__ xz,
                                                   const float* __restrict__ dtw,
                                                   const float* __restrict__ dtb,
                                                   const float* __restrict__ Alog,
                                                   const float* __restrict__ Dp,
                                                   unsigned short* __restrict__ yg, int dir) {
    __shared__ float sh[XDBL_LD];                       // one x_dbl row: [0..7]=dt_r [8..23]=B [24..39]=C
    const int tid  = threadIdx.x;
    const int lane = tid & 31;
    const int n    = lane & 15;
    const int dloc = (tid >> 5) * 2 + (lane >> 4);      // 0..15
    const int b    = blockIdx.x / 10;
    const int d    = (blockIdx.x % 10) * 16 + dloc;

    const float Adn  = -__expf(Alog[d * DS + n]);
    const float bias = dtb[d];
    const float Dd   = Dp[d];
    float wr[RK];
#pragma unroll
    for (int r = 0; r < RK; ++r) wr[r] = dtw[d * RK + r];

    float h = 0.f;
    const size_t base = (size_t)b * Lseq;
    for (int l = 0; l < Lseq; ++l) {
        __syncthreads();
        if (tid < XDBL_LD) sh[tid] = xdbl[(base + l) * XDBL_LD + tid];
        __syncthreads();
        float dl = bias;
#pragma unroll
        for (int r = 0; r < RK; ++r) dl += sh[r] * wr[r];
        const float dt = (dl > 20.f) ? dl : log1pf(__expf(dl));       // softplus
        const float xt = bf2f(xmc[(base + l) * DI + d]);
        const float dA = __expf(dt * Adn);
        h = dA * h + (dt * xt) * sh[8 + n];
        float p = h * sh[24 + n];
        p += __shfl_xor(p, 1, 16);
        p += __shfl_xor(p, 2, 16);
        p += __shfl_xor(p, 4, 16);
        p += __shfl_xor(p, 8, 16);                      // y = sum over n within 16-lane group
        if (n == 0) {
            float y = p + xt * Dd;
            float z = bf2f(xz[(base + l) * (2 * DI) + DI + d]);
            float g = z / (1.f + __expf(-z));           // silu(z) gate
            int lo  = dir ? (Lseq - 1 - l) : l;         // un-reverse on store
            yg[(base + lo) * DI + d] = f2bf(y * g);
        }
    }
}

// ---------------- out = seq + scale * fused, back to (B,C,H,W) ----------------
__global__ __launch_bounds__(256) void final_kernel(const float* __restrict__ x,
                                                    const float* __restrict__ fused,
                                                    const float* __restrict__ scale,
                                                    float* __restrict__ out) {
    int idx = blockIdx.x * 256 + threadIdx.x;           // over B*C*L, l fastest
    if (idx >= Bsz * Cch * Lseq) return;
    const int l = idx & (Lseq - 1);
    const int c = (idx >> 12) & (Cch - 1);
    const int b = idx >> 19;
    out[idx] = x[idx] + scale[0] * fused[((size_t)b * Lseq + l) * Cch + c];
}

// ---------------- host launcher ----------------
extern "C" void kernel_launch(void* const* d_in, const int* in_sizes, int n_in,
                              void* d_out, int out_size, void* d_ws, size_t ws_size,
                              hipStream_t stream) {
    (void)in_sizes; (void)n_in; (void)out_size; (void)ws_size;
    const float* x        = (const float*)d_in[0];
    const float* ln_g     = (const float*)d_in[1];
    const float* ln_b     = (const float*)d_in[2];
    const float* in_projw = (const float*)d_in[3];
    const float* conv_w   = (const float*)d_in[4];
    const float* conv_b   = (const float*)d_in[5];
    const float* x_projw  = (const float*)d_in[6];
    const float* dt_projw = (const float*)d_in[7];
    const float* dt_projb = (const float*)d_in[8];
    const float* A_log    = (const float*)d_in[9];
    const float* D_param  = (const float*)d_in[10];
    const float* out_projw= (const float*)d_in[11];
    const float* fuse_w   = (const float*)d_in[12];
    const float* scale    = (const float*)d_in[13];
    float* out = (float*)d_out;

    char*  base = (char*)d_ws;
    size_t off  = 0;
    auto alloc = [&](size_t bytes) -> void* {
        off = (off + 255) & ~(size_t)255;
        void* p = base + off; off += bytes; return p;
    };
    unsigned short* seqn = (unsigned short*)alloc((size_t)Mrows * Cch * 2);
    unsigned short* xz   = (unsigned short*)alloc((size_t)Mrows * 2 * DI * 2);
    unsigned short* xmc  = (unsigned short*)alloc((size_t)Mrows * DI * 2);
    float*          xdbl = (float*)         alloc((size_t)Mrows * XDBL_LD * 4);
    unsigned short* yg   = (unsigned short*)alloc((size_t)Mrows * DI * 2);
    unsigned short* ycat = (unsigned short*)alloc((size_t)Mrows * 2 * Cch * 2);
    float*          fusd = (float*)         alloc((size_t)Mrows * Cch * 4);
    unsigned short* w_in = (unsigned short*)alloc((size_t)2 * DI * Cch * 2);
    unsigned short* w_xp = (unsigned short*)alloc((size_t)XDBL_LD * DI * 2);
    unsigned short* w_op = (unsigned short*)alloc((size_t)Cch * DI * 2);
    unsigned short* w_fu = (unsigned short*)alloc((size_t)Cch * 2 * Cch * 2);
    unsigned short* w_cv = (unsigned short*)alloc((size_t)DI * 4 * 2);

    auto cvblk = [](int n) { return (n + 255) / 256; };
    convert_pad_bf16<<<cvblk(2*DI*Cch),     256, 0, stream>>>(in_projw,  w_in, 2*DI,    Cch,     2*DI);
    convert_pad_bf16<<<cvblk(XDBL_LD*DI),   256, 0, stream>>>(x_projw,   w_xp, RK+2*DS, DI,      XDBL_LD);
    convert_pad_bf16<<<cvblk(Cch*DI),       256, 0, stream>>>(out_projw, w_op, Cch,     DI,      Cch);
    convert_pad_bf16<<<cvblk(Cch*2*Cch),    256, 0, stream>>>(fuse_w,    w_fu, Cch,     2*Cch,   Cch);
    convert_pad_bf16<<<cvblk(DI*4),         256, 0, stream>>>(conv_w,    w_cv, DI,      4,       DI);

    layernorm_kernel<<<Mrows / 8, 256, 0, stream>>>(x, ln_g, ln_b, seqn);

    // Register-blocked tiling: wave tile = MT*16 rows x NT*16 cols; block = 8 waves in M.
    const dim3 gIn (2 * DI / 64,   Mrows / 256);   // N=320: NT=4 -> grid.x=5
    const dim3 gXp (XDBL_LD / 48,  Mrows / 256);   // N=48 : NT=3 -> grid.x=1
    const dim3 gOp (Cch / 64,      Mrows / 256);   // N=128: NT=4 -> grid.x=2
    const dim3 gFu (Cch / 64,      Mrows / 256);   // N=128: NT=4 -> grid.x=2

    for (int dir = 0; dir < 2; ++dir) {
        // in_proj: (B*L,128)x(320,128)^T -> xz (bf16), reversed rows for dir=1
        gemm_wmma<unsigned short, 2, 4><<<gIn, 256, 0, stream>>>(seqn, w_in, xz,
            /*K=*/Cch, /*lda=*/Cch, /*ldc=*/2*DI, /*off=*/0, /*rev=*/dir);
        // depthwise causal conv + SiLU
        conv_silu_kernel<<<cvblk(Bsz * Lseq * DI), 256, 0, stream>>>(xz, w_cv, conv_b, xmc);
        // x_proj: (B*L,160)x(48,160)^T -> xdbl (f32, ld 48)
        gemm_wmma<float, 2, 3><<<gXp, 256, 0, stream>>>(xmc, w_xp, xdbl,
            /*K=*/DI, /*lda=*/DI, /*ldc=*/XDBL_LD, /*off=*/0, /*rev=*/0);
        // selective scan + dt_proj(K=8) + gating; writes un-reversed
        scan_kernel<<<Bsz * 10, 256, 0, stream>>>(xdbl, xmc, xz, dt_projw, dt_projb,
                                                  A_log, D_param, yg, dir);
        // out_proj: (B*L,160)x(128,160)^T -> ycat columns [dir*128, dir*128+128)
        gemm_wmma<unsigned short, 2, 4><<<gOp, 256, 0, stream>>>(yg, w_op, ycat,
            /*K=*/DI, /*lda=*/DI, /*ldc=*/2*Cch, /*off=*/dir * Cch, /*rev=*/0);
    }

    // fuse: (B*L,256)x(128,256)^T -> fused (f32)
    gemm_wmma<float, 2, 4><<<gFu, 256, 0, stream>>>(ycat, w_fu, fusd,
        /*K=*/2*Cch, /*lda=*/2*Cch, /*ldc=*/Cch, /*off=*/0, /*rev=*/0);

    final_kernel<<<cvblk(Bsz * Cch * Lseq), 256, 0, stream>>>(x, fusd, scale, out);
}
